// MultiHeadAttention_36636071035681
// MI455X (gfx1250) — matrix operand order repaired and checked by reading, not compiled, not run on hardware
//
#include <hip/hip_runtime.h>
#include <hip/hip_bf16.h>

// MI455X / gfx1250 multi-head attention forward.
// B=4, T=2048, E=1024, H=16, D=64.
// bf16 WMMA (v_wmma_f32_16x16x32_bf16) for all GEMMs, fp32 accumulate,
// fp32 softmax/RoPE. K tiles in the attention kernel are staged into LDS with
// the gfx1250 async DMA path (global_load_async_to_lds_b128 / ASYNCcnt),
// double-buffered so the DMA overlaps softmax + PV compute.
// Workspace layout (bytes), requires ws_size >= 88 MiB:
//   [0,16M)   x in bf16                (8M elems)
//   [16M,24M) Wq,Wk,Wv,Wo in bf16      (4 x 1M elems)
//   [24M,40M) Q  bf16 [B*H, T, D] (RoPE applied)
//   [40M,56M) K  bf16 [B*H, T, D] (RoPE applied)
//   [56M,72M) V  bf16 [B*H, T, D]
//   [72M,88M) attn out bf16 [B, T, E]

#define TT 2048
#define EE 1024
#define HH 16
#define DD 64
#define BB 4

typedef __attribute__((ext_vector_type(16))) __bf16 v16bf;
typedef __attribute__((ext_vector_type(8)))  __bf16 v8bf;
typedef __attribute__((ext_vector_type(8)))  float  v8f;

__device__ __forceinline__ v8f wmma_bf16(v16bf a, v16bf b, v8f c) {
    return __builtin_amdgcn_wmma_f32_16x16x32_bf16(
        /*neg_a=*/false, a, /*neg_b=*/false, b,
        /*c_mod=*/(short)0, c, /*reuse_a=*/false, /*reuse_b=*/false);
}

// A-fragment (16Mx32K bf16): lane L holds row m=L%16; half=L/16 selects the
// k-range {half*16..half*16+15} (two 16B chunks).
__device__ __forceinline__ v16bf load_a_frag(const __bf16* rowp, int k0, int half) {
    union { v16bf v; v8bf h[2]; } u;
    u.h[0] = *(const v8bf*)(rowp + k0 + half * 16);
    u.h[1] = *(const v8bf*)(rowp + k0 + half * 16 + 8);
    return u.v;
}

// B-fragment (32Kx16N bf16): lane L holds col n=L%16, k = half*16 + i
// (16 contiguous elements = one 32B load from a row-major N x K source row).
__device__ __forceinline__ v16bf load_b_frag(const __bf16* colrow, int k0, int half) {
    return *(const v16bf*)(colrow + k0 + half * 16);
}

// ---------------------------------------------------------------------------
// fp32 -> bf16 convert
// ---------------------------------------------------------------------------
__global__ void f32_to_bf16_kernel(const float* __restrict__ in,
                                   __bf16* __restrict__ out, int n) {
    int i = blockIdx.x * blockDim.x + threadIdx.x;
    if (i < n) out[i] = (__bf16)in[i];
}

// ---------------------------------------------------------------------------
// GEMM: C[m, n] = sum_k A[m,k] * W[n,k]   (A: MxK row major, W: NxK row major)
// One wave computes a 64(M) x 64(N) tile: 4 M-subtiles share every B-fragment.
// MODE 0: RoPE epilogue, store bf16 [B*H, T, D]
// MODE 1: plain epilogue, store bf16 [B*H, T, D]
// MODE 2: plain epilogue, store f32 flat [M, E]
// ---------------------------------------------------------------------------
template <int MODE>
__global__ __launch_bounds__(32)
void gemm64x64_wmma(const __bf16* __restrict__ A, const __bf16* __restrict__ W,
                    void* __restrict__ outv) {
    const int lane  = threadIdx.x;
    const int mlane = lane & 15;
    const int halfl = lane >> 4;
    const int m0 = blockIdx.x * 64;
    const int nb = blockIdx.y;      // head index / n-block
    const int n0 = nb * 64;

    v8f acc[4][4] = {};

    const __bf16* arow[4];
#pragma unroll
    for (int mi = 0; mi < 4; ++mi)
        arow[mi] = A + (size_t)(m0 + mi * 16 + mlane) * EE;
    const __bf16* wbase = W + (size_t)(n0 + mlane) * EE;

    for (int k0 = 0; k0 < EE; k0 += 32) {
        v16bf a[4];
#pragma unroll
        for (int mi = 0; mi < 4; ++mi) a[mi] = load_a_frag(arow[mi], k0, halfl);
#pragma unroll
        for (int c = 0; c < 4; ++c) {
            v16bf b = load_b_frag(wbase + (size_t)(c * 16) * EE, k0, halfl);
#pragma unroll
            for (int mi = 0; mi < 4; ++mi)
                acc[mi][c] = wmma_bf16(a[mi], b, acc[mi][c]);
        }
    }

#pragma unroll
    for (int mi = 0; mi < 4; ++mi) {
#pragma unroll
        for (int r = 0; r < 8; ++r) {
            const int m  = m0 + mi * 16 + r + 8 * halfl;
            const int t  = m & (TT - 1);
            const int bi = m >> 11;          // m / T
#pragma unroll
            for (int c = 0; c < 4; ++c) {
                const int d = c * 16 + mlane;
                float val = acc[mi][c][r];
                if (MODE == 0) {
                    // RoPE: q*cos + rotate_half(q)*sin; partner at d +/- 32
                    // lives in fragment c^2, same lane & element.
                    float part = (c < 2) ? -acc[mi][c + 2][r] : acc[mi][c - 2][r];
                    float inv  = __powf(10000.0f, -(float)(d & 31) * (1.0f / 32.0f));
                    float ang  = (float)t * inv;
                    val = val * __cosf(ang) + part * __sinf(ang);
                }
                if (MODE == 2) {
                    ((float*)outv)[(size_t)m * EE + n0 + d] = val;
                } else {
                    // [B*H, T, D] with bh = bi*H + nb
                    ((__bf16*)outv)[(((size_t)bi * HH + nb) * TT + t) * DD + d] =
                        (__bf16)val;
                }
            }
        }
    }
}

// ---------------------------------------------------------------------------
// Flash attention: 4 waves / block, wave w owns q rows [qb0+16w, qb0+16w+16).
// 64 keys per iteration. K tiles (contiguous 8KB) are DMA'd into LDS with
// global_load_async_to_lds_b128, double-buffered: the prefetch of slab kb+64
// is issued right after slab kb's S-tiles are consumed, overlapping the DMA
// with softmax VALU work and the PV WMMAs. V tile staged transposed in LDS.
// Block loop bound is uniform (kb <= qb0), so __syncthreads is safe.
// ---------------------------------------------------------------------------
__global__ __launch_bounds__(128)
void flash_attn_wmma(const __bf16* __restrict__ Q, const __bf16* __restrict__ Kb,
                     const __bf16* __restrict__ Vb, __bf16* __restrict__ O) {
    __shared__ __bf16 kt[2][64][64];     // double-buffered K tiles (DMA) 16KB
    __shared__ __bf16 vt[64][64];        // V tile transposed: vt[d][key]  8KB
    __shared__ __bf16 pst[4][16][64];    // per-wave P tiles (C->A repack) 8KB

    const int tid   = threadIdx.x;
    const int lane  = tid & 31;
    const int w     = tid >> 5;
    const int mlane = lane & 15;
    const int halfl = lane >> 4;
    const int bh  = blockIdx.y;          // b*H + h
    const int qb0 = blockIdx.x * 64;     // block q base
    const int q0  = qb0 + w * 16;        // wave q base

    const __bf16* qp = Q  + ((size_t)bh * TT + q0) * DD;
    const __bf16* kp = Kb + (size_t)bh * TT * DD;
    const __bf16* vp = Vb + (size_t)bh * TT * DD;

    const unsigned ktoff0 = (unsigned)(uintptr_t)&kt[0][0][0];
    const unsigned ktoff1 = (unsigned)(uintptr_t)&kt[1][0][0];

    // Issue async DMA of one 8KB K tile (rows kbase..kbase+63) into buffer.
    // 128 threads x 4 x b128 = 8KB; linear copy (rows are 128B contiguous).
    auto issue_k_tile = [&](int kbase, int buf) {
        unsigned long long g = (unsigned long long)(uintptr_t)(kp + (size_t)kbase * DD)
                               + (unsigned)(tid * 16);
        unsigned l = (buf ? ktoff1 : ktoff0) + (unsigned)(tid * 16);
#pragma unroll
        for (int i = 0; i < 4; ++i) {
            asm volatile("global_load_async_to_lds_b128 %0, %1, off"
                         :: "v"(l + (unsigned)(i * 2048)),
                            "v"(g + (unsigned long long)(i * 2048))
                         : "memory");
        }
    };

    // Q A-fragments (16x64 -> two 16x32 chunks), live for the whole loop.
    v16bf qa[2];
    qa[0] = load_a_frag(qp + (size_t)mlane * DD, 0, halfl);
    qa[1] = load_a_frag(qp + (size_t)mlane * DD, 32, halfl);

    v8f o[4] = {};
    float rowmax[8], rowsum[8];
#pragma unroll
    for (int r = 0; r < 8; ++r) { rowmax[r] = -1e30f; rowsum[r] = 0.0f; }

    issue_k_tile(0, 0);

    for (int kb = 0; kb <= qb0; kb += 64) {
        const int buf = (kb >> 6) & 1;
        // Wait for this slab's K DMA (own counter), publish to all waves.
        asm volatile("s_wait_asynccnt 0x0" ::: "memory");
        __syncthreads();

        // ---- S = Q @ K^T for 64 keys (4 n-tiles); K from LDS ----
        float x[4][8];
#pragma unroll
        for (int nt = 0; nt < 4; ++nt) {
            if (kb + nt * 16 <= q0 + 15) {           // wave-uniform branch
                v8f s = {};
                const __bf16* krow = &kt[buf][nt * 16 + mlane][0];
#pragma unroll
                for (int ch = 0; ch < 2; ++ch) {
                    v16bf b = load_b_frag(krow, ch * 32, halfl);
                    s = wmma_bf16(qa[ch], b, s);
                }
#pragma unroll
                for (int r = 0; r < 8; ++r) {
                    const int qrow = q0 + r + 8 * halfl;
                    const int kcol = kb + nt * 16 + mlane;
                    float xv = s[r] * 0.125f + ((kcol > qrow) ? -10000.0f : 0.0f);
                    x[nt][r] = fminf(fmaxf(xv, -10000.0f), 10000.0f);
                }
            } else {
#pragma unroll
                for (int r = 0; r < 8; ++r) x[nt][r] = -10000.0f;
            }
        }

        // Prefetch next slab's K tile into the other buffer; overlaps with
        // the softmax VALU work and PV WMMAs below.
        if (kb + 64 <= qb0) issue_k_tile(kb + 64, buf ^ 1);

        // ---- online softmax over the 64-key slab (fp32) ----
#pragma unroll
        for (int r = 0; r < 8; ++r) {
            float mx = fmaxf(fmaxf(x[0][r], x[1][r]), fmaxf(x[2][r], x[3][r]));
#pragma unroll
            for (int off = 1; off < 16; off <<= 1)
                mx = fmaxf(mx, __shfl_xor(mx, off, 16));
            const float nm = fmaxf(rowmax[r], mx);
            const float al = __expf(rowmax[r] - nm);
            rowmax[r] = nm;

            float ps = 0.0f;
#pragma unroll
            for (int nt = 0; nt < 4; ++nt) {
                x[nt][r] = __expf(x[nt][r] - nm);    // x becomes P
                ps += x[nt][r];
            }
#pragma unroll
            for (int off = 1; off < 16; off <<= 1)
                ps += __shfl_xor(ps, off, 16);
            rowsum[r] = rowsum[r] * al + ps;

#pragma unroll
            for (int c = 0; c < 4; ++c) o[c][r] = o[c][r] * al;
        }

        // ---- stage P (C layout -> row-major) and V (transposed) in LDS ----
        __syncthreads();                 // previous iter's vt/pst readers done
#pragma unroll
        for (int r = 0; r < 8; ++r) {
            const int pr = r + 8 * halfl;
#pragma unroll
            for (int nt = 0; nt < 4; ++nt)
                pst[w][pr][nt * 16 + mlane] = (__bf16)x[nt][r];
        }
        {
            // 128 threads stage 64 keys x 64 d: key = tid/2, d-half = tid&1.
            const int key = tid >> 1;
            const int dh  = (tid & 1) * 32;
            const v8bf* src = (const v8bf*)(vp + (size_t)(kb + key) * DD + dh);
#pragma unroll
            for (int i = 0; i < 4; ++i) {
                v8bf chunk = src[i];
#pragma unroll
                for (int j = 0; j < 8; ++j) vt[dh + i * 8 + j][key] = chunk[j];
            }
        }
        __syncthreads();

        // ---- O += P @ V  (K = 64 keys, two 32-key chunks) ----
#pragma unroll
        for (int ch = 0; ch < 2; ++ch) {
            v16bf pa = load_a_frag(&pst[w][mlane][0], ch * 32, halfl);
#pragma unroll
            for (int c = 0; c < 4; ++c) {
                v16bf vb = *(const v16bf*)(&vt[c * 16 + mlane][ch * 32 + halfl * 16]);
                o[c] = wmma_bf16(pa, vb, o[c]);
            }
        }
    }

    // ---- normalize and store bf16 [B, T, E] ----
    const int bi = bh >> 4;
    const int h  = bh & 15;
#pragma unroll
    for (int r = 0; r < 8; ++r) {
        const int t = q0 + r + 8 * halfl;
        const float inv = 1.0f / rowsum[r];
#pragma unroll
        for (int c = 0; c < 4; ++c) {
            const int e = h * 64 + c * 16 + mlane;
            O[((size_t)bi * TT + t) * EE + e] = (__bf16)(o[c][r] * inv);
        }
    }
}

// ---------------------------------------------------------------------------
extern "C" void kernel_launch(void* const* d_in, const int* in_sizes, int n_in,
                              void* d_out, int out_size, void* d_ws, size_t ws_size,
                              hipStream_t stream) {
    (void)in_sizes; (void)n_in; (void)out_size; (void)ws_size;
    const float* x  = (const float*)d_in[0];
    // d_in[1] = causal mask (recomputed analytically in-kernel)
    const float* Wq = (const float*)d_in[2];
    const float* Wk = (const float*)d_in[3];
    const float* Wv = (const float*)d_in[4];
    const float* Wo = (const float*)d_in[5];

    char* ws = (char*)d_ws;
    __bf16* xb  = (__bf16*)ws;                               // 8M elems
    __bf16* wqb = (__bf16*)(ws + (size_t)(16 << 20));        // 1M
    __bf16* wkb = wqb + (1 << 20);
    __bf16* wvb = wkb + (1 << 20);
    __bf16* wob = wvb + (1 << 20);
    __bf16* qb  = (__bf16*)(ws + (size_t)(24 << 20));        // 8M
    __bf16* kb  = qb + (size_t)(8 << 20);
    __bf16* vb  = kb + (size_t)(8 << 20);
    __bf16* ab  = vb + (size_t)(8 << 20);

    const int NX = BB * TT * EE;        // 8M
    const int NW = EE * EE;             // 1M
    f32_to_bf16_kernel<<<(NX + 255) / 256, 256, 0, stream>>>(x,  xb,  NX);
    f32_to_bf16_kernel<<<(NW + 255) / 256, 256, 0, stream>>>(Wq, wqb, NW);
    f32_to_bf16_kernel<<<(NW + 255) / 256, 256, 0, stream>>>(Wk, wkb, NW);
    f32_to_bf16_kernel<<<(NW + 255) / 256, 256, 0, stream>>>(Wv, wvb, NW);
    f32_to_bf16_kernel<<<(NW + 255) / 256, 256, 0, stream>>>(Wo, wob, NW);

    dim3 gg((BB * TT) / 64, HH);        // 128 x 16
    gemm64x64_wmma<0><<<gg, 32, 0, stream>>>(xb, wqb, (void*)qb);  // Q + RoPE
    gemm64x64_wmma<0><<<gg, 32, 0, stream>>>(xb, wkb, (void*)kb);  // K + RoPE
    gemm64x64_wmma<1><<<gg, 32, 0, stream>>>(xb, wvb, (void*)vb);  // V

    flash_attn_wmma<<<dim3(TT / 64, BB * HH), 128, 0, stream>>>(qb, kb, vb, ab);

    gemm64x64_wmma<2><<<gg, 32, 0, stream>>>(ab, wob, d_out);      // O-proj
}
